// DSAIndexer_59777354826253
// MI455X (gfx1250) — matrix operand order repaired and checked by reading, not compiled, not run on hardware
//
#include <hip/hip_runtime.h>
#include <hip/hip_bf16.h>

typedef __attribute__((ext_vector_type(16))) __bf16 v16bf;
typedef __attribute__((ext_vector_type(8)))  float  v8f;

#define SQ   4096
#define SK   4096
#define NH   16
#define HD   128
#define TOPK 2048

// ---------------------------------------------------------------------------
// Kernel 0: one-shot K pre-conversion f32 -> bf16, stored in the exact
// per-lane WMMA B-fragment layout: frag[(chunk*4 + kstep)*32 + lane][16 bf16].
// B-tile (32d x 16keys): lane = key column, 2 K-values per VGPR,
// lanes 16-31 hold K+16.  Total: 256 chunks * 4 ksteps * 32 lanes * 16 = 1 MB.
// ---------------------------------------------------------------------------
__global__ __launch_bounds__(256)
void dsa_convert_k_kernel(const float* __restrict__ k_ptr,
                          __bf16* __restrict__ kfrag) {
  const int idx = blockIdx.x * blockDim.x + threadIdx.x;  // one bf16 pair each
  if (idx >= (SK / 16) * 4 * 32 * 8) return;              // 262144 pairs
  const int p    = idx & 7;          // pair index within lane fragment
  const int lane = (idx >> 3) & 31;
  const int c    = (idx >> 8) & 3;   // k-step (d chunk of 32)
  const int ch   = idx >> 10;        // key chunk of 16
  const int half = lane >> 4, l16 = lane & 15;
  const int key  = ch * 16 + l16;
  const int d    = c * 32 + half * 16 + 2 * p;
  const float2 f = *(const float2*)(k_ptr + (size_t)key * HD + d);
  kfrag[2 * idx]     = (__bf16)f.x;
  kfrag[2 * idx + 1] = (__bf16)f.y;
}

// ---------------------------------------------------------------------------
// Kernel 1: scores[q,k] = sum_h w[q,h] * relu( (Q[q,h,:]·K[k,:]) / sqrt(128) )
// with causal mask (k > q -> -inf). One wave32 per query row.
// A-tile = the query's 16 heads x 128 dims, converted to bf16 once per wave.
// Inner loop: 64 keys / iteration = 16 v_wmma_f32_16x16x32_bf16 across 4
// independent accumulator chains (k-step outer, chunk inner) so B-loads for
// step c+1 overlap the WMMAs of step c.
// ---------------------------------------------------------------------------
__global__ __launch_bounds__(256)
void dsa_scores_kernel(const float* __restrict__ q_ptr,
                       const v16bf* __restrict__ kfrag,
                       const float* __restrict__ w_ptr,
                       float* __restrict__ scores_out) {
  const int wave = blockIdx.x * (blockDim.x >> 5) + (threadIdx.x >> 5);
  if (wave >= SQ) return;
  const int q    = wave;
  const int lane = threadIdx.x & 31;
  const int half = lane >> 4;   // 0: lanes 0-15, 1: lanes 16-31
  const int l16  = lane & 15;

  // ---- A fragments: 16 heads x 128 dims, bf16, 4 k-steps of K=32 ---------
  // A k order used here, contiguous, not AMD's table order (matches kfrag): lane(0-15)=row M, vgpr v -> K=2v,2v+1 ;
  // lanes 16-31 same rows with K offset +16.
  const float* qbase = q_ptr + (size_t)q * NH * HD;   // 16x128 row-major
  v16bf afrag[4];
  #pragma unroll
  for (int c = 0; c < 4; ++c) {
    #pragma unroll
    for (int v = 0; v < 8; ++v) {
      const int d = c * 32 + 2 * v + half * 16;
      const float2 f = *(const float2*)(qbase + l16 * HD + d);
      afrag[c][2 * v]     = (__bf16)f.x;
      afrag[c][2 * v + 1] = (__bf16)f.y;
    }
  }

  // Per-lane head weights for the accumulator rows this lane reduces:
  // C/D layout: VGPR r holds M = r (lanes 0-15) / M = r+8 (lanes 16-31).
  float wv[8];
  #pragma unroll
  for (int r = 0; r < 8; ++r)
    wv[r] = w_ptr[(size_t)q * NH + r + half * 8];

  const float scale = 0.08838834764831845f;  // 1/sqrt(128)
  float* row_out = scores_out + (size_t)q * SK;
  const int ngroups = (q >> 6) + 1;          // 64-key groups in causal range

  for (int g = 0; g < SK / 64; ++g) {
    const int ch0 = g * 4;                   // first 16-key chunk of group
    if (g < ngroups) {
      const v16bf* kf = kfrag + (size_t)ch0 * 4 * 32;
      // Uniform speculative prefetch of the next group's fragments.
      __builtin_prefetch(kfrag + (size_t)(ch0 + 4) * 4 * 32 + lane, 0, 3);

      v8f acc[4] = {v8f{}, v8f{}, v8f{}, v8f{}};
      #pragma unroll
      for (int c = 0; c < 4; ++c) {          // k-step (d chunk of 32)
        #pragma unroll
        for (int cc = 0; cc < 4; ++cc) {     // key chunk within group
          const v16bf b = kf[(cc * 4 + c) * 32 + lane];
          acc[cc] = __builtin_amdgcn_wmma_f32_16x16x32_bf16(
              false, afrag[c], false, b, (short)0, acc[cc], false, false);
        }
      }
      // relu + head-weighted reduction over M (heads), merge half-rows.
      #pragma unroll
      for (int cc = 0; cc < 4; ++cc) {
        float s = 0.f;
        #pragma unroll
        for (int r = 0; r < 8; ++r) {
          float v = acc[cc][r];
          v = v > 0.f ? v : 0.f;
          s += v * wv[r];
        }
        s += __shfl_xor(s, 16, 32);          // lane L <-> L^16: full head sum
        if (half == 0) {
          const int kk = (ch0 + cc) * 16 + l16;
          row_out[kk] = (kk <= q) ? s * scale : -__builtin_inff();
        }
      }
    } else if (half == 0) {
      #pragma unroll
      for (int cc = 0; cc < 4; ++cc)
        row_out[(ch0 + cc) * 16 + l16] = -__builtin_inff();
    }
  }
}

// ---------------------------------------------------------------------------
// Kernel 2: per-row top-2048 via full bitonic sort of 4096 (val,idx) pairs in
// LDS (32 KB of the 320 KB WGP budget). Comparator = (value desc, index asc),
// matching lax.top_k ordering/tie-breaking. One 1024-thread block per row.
// ---------------------------------------------------------------------------
__global__ __launch_bounds__(1024)
void dsa_topk_kernel(const float* __restrict__ scores,
                     int* __restrict__ out_idx) {
  __shared__ float sval[SK];
  __shared__ int   sidx[SK];
  const int q = blockIdx.x;
  const float* row = scores + (size_t)q * SK;
  for (int i = threadIdx.x; i < SK; i += blockDim.x) {
    sval[i] = row[i];
    sidx[i] = i;
  }
  __syncthreads();

  for (int ksz = 2; ksz <= SK; ksz <<= 1) {
    for (int j = ksz >> 1; j >= 1; j >>= 1) {
      for (int t = threadIdx.x; t < SK / 2; t += blockDim.x) {
        const int i   = ((t & ~(j - 1)) << 1) | (t & (j - 1));
        const int ixj = i + j;
        const float vi = sval[i], vj = sval[ixj];
        const int   ii = sidx[i], ij = sidx[ixj];
        // "j comes first" under (val desc, idx asc); keys unique via idx.
        const bool jFirst = (vj > vi) || (vj == vi && ij < ii);
        const bool dir    = ((i & ksz) == 0);
        if (dir == jFirst) {
          sval[i] = vj; sval[ixj] = vi;
          sidx[i] = ij; sidx[ixj] = ii;
        }
      }
      __syncthreads();
    }
  }

  for (int i = threadIdx.x; i < TOPK; i += blockDim.x)
    out_idx[(size_t)q * TOPK + i] = sidx[i];
}

// ---------------------------------------------------------------------------
extern "C" void kernel_launch(void* const* d_in, const int* in_sizes, int n_in,
                              void* d_out, int out_size, void* d_ws, size_t ws_size,
                              hipStream_t stream) {
  (void)in_sizes; (void)n_in; (void)out_size; (void)ws_size;
  const float* index_q = (const float*)d_in[0];   // [1,4096,16,128] f32
  const float* index_k = (const float*)d_in[1];   // [1,4096,128]   f32
  const float* index_w = (const float*)d_in[2];   // [1,4096,16]    f32
  // d_in[3] = attn_mask (causal tril, applied analytically)
  // d_in[4] = topk scalar (2048, compile-time constant here)

  // d_out layout: [topk_indices int32 4096*2048][scores f32 4096*4096]
  int*   idx_out    = (int*)d_out;
  float* scores_out = (float*)d_out + (size_t)SQ * TOPK;

  // d_ws: 1 MB bf16 K in WMMA B-fragment order.
  __bf16* kfrag = (__bf16*)d_ws;

  // Kernel 0: 262144 bf16 pairs, one per thread.
  dsa_convert_k_kernel<<<1024, 256, 0, stream>>>(index_k, kfrag);

  // 4096 waves (one per query), 8 waves per 256-thread block.
  dsa_scores_kernel<<<SQ / 8, 256, 0, stream>>>(index_q, (const v16bf*)kfrag,
                                                index_w, scores_out);

  // One block per query row for the sort/top-k.
  dsa_topk_kernel<<<SQ, 1024, 0, stream>>>(scores_out, idx_out);
}
